// SolvGNNV5_37778532335675
// MI455X (gfx1250) — matrix operand order repaired and checked by reading, not compiled, not run on hardware
//
#include <hip/hip_runtime.h>

typedef __attribute__((ext_vector_type(16))) _Float16 v16h;
typedef __attribute__((ext_vector_type(8)))  _Float16 v8h;
typedef __attribute__((ext_vector_type(8)))  float    v8f;
typedef __attribute__((ext_vector_type(4)))  unsigned int v4u;
typedef __attribute__((ext_vector_type(8)))  int v8i;
typedef __attribute__((ext_vector_type(4)))  int v4i;

#define NODES     20000
#define NODES_PAD 20032   // 313 * 64, so GEMM blocks need no tail predication
#define EDGES     320000
#define GRAPHS    64
#define INDIM     74
#define KPRE      96      // INDIM padded to a multiple of 32 for WMMA K-steps
#define HID       256
#define NGIN      24

// ---------------- elementwise helpers ----------------
__global__ void copy_f32(const float* __restrict__ s, float* __restrict__ d, int n) {
  int i = blockIdx.x * blockDim.x + threadIdx.x;
  if (i < n) d[i] = s[i];
}

__global__ void zero_f32(float* __restrict__ p, int n) {
  int i = blockIdx.x * blockDim.x + threadIdx.x;
  if (i < n) p[i] = 0.f;
}

// t [NODES][HID] f32 -> A [NODES_PAD][HID] f16 (zero pad rows)
__global__ void cvt_a256(const float* __restrict__ t, _Float16* __restrict__ A) {
  int i = blockIdx.x * blockDim.x + threadIdx.x;
  if (i >= NODES_PAD * HID) return;
  A[i] = (i < NODES * HID) ? (_Float16)t[i] : (_Float16)0.f;
}

// t [NODES][INDIM] f32 -> A [NODES_PAD][KPRE] f16 (zero pad cols+rows)
__global__ void cvt_a_pre(const float* __restrict__ t, _Float16* __restrict__ A) {
  int i = blockIdx.x * blockDim.x + threadIdx.x;
  if (i >= NODES_PAD * KPRE) return;
  int node = i / KPRE, k = i - node * KPRE;
  A[i] = (node < NODES && k < INDIM) ? (_Float16)t[node * INDIM + k] : (_Float16)0.f;
}

// W_pre [INDIM][HID] f32 -> transposed+padded f16 [HID][KPRE]
__global__ void cvt_wpre(const float* __restrict__ W, _Float16* __restrict__ Wt) {
  int i = blockIdx.x * blockDim.x + threadIdx.x;
  if (i >= HID * KPRE) return;
  int nn = i / KPRE, k = i - nn * KPRE;
  Wt[i] = (k < INDIM) ? (_Float16)W[k * HID + nn] : (_Float16)0.f;
}

// W_stack [L][K][N] f32 -> transposed f16 [L][N][K]
__global__ void cvt_wstack(const float* __restrict__ W, _Float16* __restrict__ Wt, int n) {
  int i = blockIdx.x * blockDim.x + threadIdx.x;
  if (i >= n) return;
  int l = i >> 16, rem = i & 65535, nn = rem >> 8, k = rem & 255;
  Wt[i] = (_Float16)W[(l << 16) + (k << 8) + nn];
}

// ---------------- aggregation (scatter-add into t, L2-resident) ----------------
__global__ void edge_scatter_h(const float* __restrict__ h, const int* __restrict__ src,
                               const int* __restrict__ dst, float* __restrict__ t) {
  long long tid = (long long)blockIdx.x * blockDim.x + threadIdx.x;
  int e = (int)(tid >> 6);
  if (e >= EDGES) return;
  int c = ((int)tid & 63) << 2;                       // 4 channels/thread, coalesced
  const float4 v = *(const float4*)(h + (size_t)src[e] * HID + c);
  float* p = t + (size_t)dst[e] * HID + c;
  atomicAdd(p + 0, v.x); atomicAdd(p + 1, v.y);
  atomicAdd(p + 2, v.z); atomicAdd(p + 3, v.w);
}

__global__ void edge_scatter_pre(const float* __restrict__ x, const int* __restrict__ src,
                                 const int* __restrict__ dst, float* __restrict__ t) {
  long long tid = (long long)blockIdx.x * blockDim.x + threadIdx.x;
  int e = (int)(tid >> 7);
  int c = (int)tid & 127;
  if (e >= EDGES || c >= INDIM) return;
  atomicAdd(t + (size_t)dst[e] * INDIM + c, x[(size_t)src[e] * INDIM + c]);
}

// ---------------- WMMA GEMM: out[Mpad][256] = relu(A[Mpad][K] @ Wt^T + bias) ----------
// A: f16 row-major [Mpad][K]; Wt: f16 [256][K] (W transposed); K in {96, 256} (template).
// Weight tile staged into LDS by the Tensor Data Mover (one 2D D# descriptor, wave 0),
// falling back to a cooperative vector copy if the TDM builtin is unavailable.
// Block = 8 waves; wave (wm,wn) computes 32(M) x 64(N): 2 A-frags x 4 B-frags -> 8 WMMAs
// per k-step; k-loop fully unrolled (K compile-time) so LLVM schedules loads across steps.
union AF { v16h v; v8h h[2]; };

__device__ inline void store_tile(float* __restrict__ out, const v8f& acc,
                                  const float* __restrict__ bias, int rbase, int ncol) {
  float bv = bias[ncol];
#pragma unroll
  for (int r = 0; r < 8; ++r) {
    float v = acc[r] + bv;
    out[(size_t)(rbase + r) * HID + ncol] = v > 0.f ? v : 0.f;
  }
}

template <int K>
__global__ __launch_bounds__(256)
void gemm_wmma_relu(const _Float16* __restrict__ A, const _Float16* __restrict__ Wt,
                    const float* __restrict__ bias, float* __restrict__ out) {
  __shared__ __align__(32) _Float16 lds[HID * K];     // weight tile, [N][K] layout

#if __has_builtin(__builtin_amdgcn_tensor_load_to_lds) && __has_builtin(__builtin_amdgcn_s_wait_tensorcnt)
  if (threadIdx.x < 32) {                             // wave 0 drives the TDM
    const unsigned long long ga = (unsigned long long)Wt;
    const unsigned int lds_off = (unsigned int)(unsigned long long)&lds[0];
    // D# group 0: count=1 | lds_addr | global_addr[56:0] | type=2
    v4u g0 = { 1u, lds_off, (unsigned int)(ga & 0xffffffffu),
               (unsigned int)((ga >> 32) & 0x1ffffffu) | (2u << 30) };
    // D# group 1: data_size=1 (2B); tensor 2D: dim0=K, dim1=HID; tile = K x HID; stride0=K
    v8i g1 = { (int)(1u << 16),                       // mask=0, data_size=2B
               (int)((unsigned)(K & 0xffff) << 16),   // tensor_dim0[15:0] @ bits 63:48
               (int)((unsigned)HID << 16),            // tensor_dim0[31:16]=0, tensor_dim1[15:0]
               (int)((unsigned)K << 16),              // tensor_dim1[31:16]=0, tile_dim0=K
               (int)HID,                              // tile_dim1=HID, tile_dim2=0
               (int)K,                                // tensor_dim0_stride[31:0]
               0, 0 };                                // stride0[47:32], stride1
    v4i gz4 = { 0, 0, 0, 0 };                         // groups 2/3 unused (2D tensor)
    v8i gz8 = { 0, 0, 0, 0, 0, 0, 0, 0 };             // extra group (6-arg toolchain form)
    __builtin_amdgcn_tensor_load_to_lds(g0, g1, gz4, gz4, gz8, 0);
    __builtin_amdgcn_s_wait_tensorcnt(0);
  }
  __syncthreads();
#else
  for (int i = threadIdx.x * 16; i < HID * K; i += 256 * 16)
    *(v16h*)(lds + i) = *(const v16h*)(Wt + i);
  __syncthreads();
#endif

  const int lane = threadIdx.x & 31;
  const int wave = threadIdx.x >> 5;
  const int m0 = blockIdx.x * 64 + (wave & 1) * 32;
  const int n0 = (wave >> 1) * 64;
  const int l15 = lane & 15;
  const int hi  = lane >> 4;
  const int aoff = hi * 16;             // A-fragment k sub-offset (contiguous k = 16*hi + s)
  const int boff = hi * 16;             // B-fragment k sub-offset (contiguous k = 16*hi + s)

  const _Float16* __restrict__ Arow0 = A + (size_t)(m0 + l15) * K + aoff;
  const _Float16* __restrict__ Arow1 = A + (size_t)(m0 + 16 + l15) * K + aoff;
  const _Float16* Blds = lds + (size_t)(n0 + l15) * K + boff;
  __builtin_prefetch(Arow0, 0, 3);
  __builtin_prefetch(Arow1, 0, 3);

  v8f acc[8] = {};                      // acc[2*f + mrow]
#pragma unroll
  for (int ks = 0; ks < K / 32; ++ks) {
    const int kb = ks * 32;
    AF a0, a1;
    a0.h[0] = *(const v8h*)(Arow0 + kb);  a0.h[1] = *(const v8h*)(Arow0 + kb + 8);
    a1.h[0] = *(const v8h*)(Arow1 + kb);  a1.h[1] = *(const v8h*)(Arow1 + kb + 8);
    v16h b0 = *(const v16h*)(Blds + 0 * 16 * K + kb);
    v16h b1 = *(const v16h*)(Blds + 1 * 16 * K + kb);
    v16h b2 = *(const v16h*)(Blds + 2 * 16 * K + kb);
    v16h b3 = *(const v16h*)(Blds + 3 * 16 * K + kb);
    acc[0] = __builtin_amdgcn_wmma_f32_16x16x32_f16(false, a0.v, false, b0, (short)0, acc[0], false, false);
    acc[1] = __builtin_amdgcn_wmma_f32_16x16x32_f16(false, a1.v, false, b0, (short)0, acc[1], false, false);
    acc[2] = __builtin_amdgcn_wmma_f32_16x16x32_f16(false, a0.v, false, b1, (short)0, acc[2], false, false);
    acc[3] = __builtin_amdgcn_wmma_f32_16x16x32_f16(false, a1.v, false, b1, (short)0, acc[3], false, false);
    acc[4] = __builtin_amdgcn_wmma_f32_16x16x32_f16(false, a0.v, false, b2, (short)0, acc[4], false, false);
    acc[5] = __builtin_amdgcn_wmma_f32_16x16x32_f16(false, a1.v, false, b2, (short)0, acc[5], false, false);
    acc[6] = __builtin_amdgcn_wmma_f32_16x16x32_f16(false, a0.v, false, b3, (short)0, acc[6], false, false);
    acc[7] = __builtin_amdgcn_wmma_f32_16x16x32_f16(false, a1.v, false, b3, (short)0, acc[7], false, false);
  }

  const int rb0 = m0 + hi * 8;          // ISA f32 C/D layout: rows r + 8*hi, col = lane&15
  const int rb1 = m0 + 16 + hi * 8;
#pragma unroll
  for (int f = 0; f < 4; ++f) {
    store_tile(out, acc[2 * f],     bias, rb0, n0 + 16 * f + l15);
    store_tile(out, acc[2 * f + 1], bias, rb1, n0 + 16 * f + l15);
  }
}

// ---------------- readout + MLP ----------------
__global__ void readout_scatter(const float* __restrict__ h, const int* __restrict__ gid,
                                float* __restrict__ gsum, float* __restrict__ cnt) {
  int i = blockIdx.x * blockDim.x + threadIdx.x;
  if (i >= NODES * HID) return;
  int node = i >> 8, c = i & 255;
  int g = gid[node];
  atomicAdd(&gsum[(g << 8) + c], h[i]);
  if (c == 0) atomicAdd(&cnt[g], 1.0f);
}

__global__ void readout_mean(const float* __restrict__ gsum, const float* __restrict__ cnt,
                             float* __restrict__ mean) {
  int i = blockIdx.x * blockDim.x + threadIdx.x;
  if (i >= GRAPHS * HID) return;
  mean[i] = gsum[i] / fmaxf(cnt[i >> 8], 1.0f);
}

__global__ void mlp(const float* __restrict__ X, const float* __restrict__ W,
                    const float* __restrict__ b, float* __restrict__ Y,
                    int K, int N, int act) {
  int i = blockIdx.x * blockDim.x + threadIdx.x;
  if (i >= GRAPHS * N) return;
  int r = i / N, n = i - r * N;
  const float* x = X + (size_t)r * K;
  float s = b[n];
  for (int k = 0; k < K; ++k) s = fmaf(x[k], W[(size_t)k * N + n], s);
  if (act) s = s > 0.f ? s : 0.01f * s;   // LeakyReLU
  Y[i] = s;
}

// ---------------- host-side orchestration ----------------
extern "C" void kernel_launch(void* const* d_in, const int* in_sizes, int n_in,
                              void* d_out, int out_size, void* d_ws, size_t ws_size,
                              hipStream_t stream) {
  const float* node_feat = (const float*)d_in[0];
  const int*   src       = (const int*)d_in[1];
  const int*   dst       = (const int*)d_in[2];
  const int*   gid       = (const int*)d_in[3];
  const float* W_pre     = (const float*)d_in[4];
  const float* b_pre     = (const float*)d_in[5];
  const float* W_stack   = (const float*)d_in[6];
  const float* b_stack   = (const float*)d_in[7];
  const float* W1 = (const float*)d_in[8];
  const float* b1 = (const float*)d_in[9];
  const float* W2 = (const float*)d_in[10];
  const float* b2 = (const float*)d_in[11];
  const float* W3 = (const float*)d_in[12];
  const float* b3 = (const float*)d_in[13];
  (void)in_sizes; (void)n_in; (void)out_size; (void)ws_size;

  char* ws = (char*)d_ws;
  size_t off = 0;
  auto carve = [&](size_t bytes) -> void* {
    void* p = ws + off; off += (bytes + 255) & ~(size_t)255; return p;
  };
  float*    t    = (float*)   carve((size_t)NODES * HID * 4);
  float*    h    = (float*)   carve((size_t)NODES_PAD * HID * 4);
  _Float16* A    = (_Float16*)carve((size_t)NODES_PAD * HID * 2);
  _Float16* WtS  = (_Float16*)carve((size_t)NGIN * HID * HID * 2);
  _Float16* WtP  = (_Float16*)carve((size_t)HID * KPRE * 2);
  float*    gsum = (float*)   carve((size_t)GRAPHS * HID * 4);
  float*    cnt  = (float*)   carve((size_t)GRAPHS * 4);
  float*    mean = (float*)   carve((size_t)GRAPHS * HID * 4);
  float*    x1   = (float*)   carve((size_t)GRAPHS * 1024 * 4);
  float*    x2   = (float*)   carve((size_t)GRAPHS * 512 * 4);

  const int B = 256;
  auto nb = [](long long n, int b) { return (int)((n + b - 1) / b); };
  const int GEMM_BLOCKS = NODES_PAD / 64;   // 313

  // weight prep (f32 -> transposed f16) — deterministic, redone every call
  cvt_wpre  <<<nb(HID * KPRE, B), B, 0, stream>>>(W_pre, WtP);
  cvt_wstack<<<nb((long long)NGIN * HID * HID, B), B, 0, stream>>>(W_stack, WtS, NGIN * HID * HID);

  // pre layer: h = relu((x + agg(x)) @ W_pre + b_pre)
  copy_f32        <<<nb(NODES * INDIM, B), B, 0, stream>>>(node_feat, t, NODES * INDIM);
  edge_scatter_pre<<<nb((long long)EDGES * 128, B), B, 0, stream>>>(node_feat, src, dst, t);
  cvt_a_pre       <<<nb((long long)NODES_PAD * KPRE, B), B, 0, stream>>>(t, A);
  gemm_wmma_relu<KPRE><<<GEMM_BLOCKS, 256, 0, stream>>>(A, WtP, b_pre, h);

  // 24 GIN layers
  for (int l = 0; l < NGIN; ++l) {
    copy_f32      <<<nb(NODES * HID, B), B, 0, stream>>>(h, t, NODES * HID);
    edge_scatter_h<<<nb((long long)EDGES * 64, B), B, 0, stream>>>(h, src, dst, t);
    cvt_a256      <<<nb((long long)NODES_PAD * HID, B), B, 0, stream>>>(t, A);
    gemm_wmma_relu<HID><<<GEMM_BLOCKS, 256, 0, stream>>>(A, WtS + (size_t)l * HID * HID,
                                                         b_stack + l * HID, h);
  }

  // per-graph mean readout
  zero_f32       <<<nb(GRAPHS * HID, B), B, 0, stream>>>(gsum, GRAPHS * HID);
  zero_f32       <<<1, 64, 0, stream>>>(cnt, GRAPHS);
  readout_scatter<<<nb(NODES * HID, B), B, 0, stream>>>(h, gid, gsum, cnt);
  readout_mean   <<<nb(GRAPHS * HID, B), B, 0, stream>>>(gsum, cnt, mean);

  // tiny MLP head (64 rows) — scalar f32
  mlp<<<nb(GRAPHS * 1024, B), B, 0, stream>>>(mean, W1, b1, x1, HID, 1024, 1);
  mlp<<<nb(GRAPHS * 512, B), B, 0, stream>>>(x1, W2, b2, x2, 1024, 512, 1);
  mlp<<<1, 64, 0, stream>>>(x2, W3, b3, (float*)d_out, 512, 1, 0);
}
